// SelfLSTM_25675314495801
// MI455X (gfx1250) — matrix operand order repaired and checked by reading, not compiled, not run on hardware
//
#include <hip/hip_runtime.h>
#include <hip/hip_bf16.h>

typedef __bf16 bf16;
typedef __attribute__((ext_vector_type(16))) __bf16 v16bf;
typedef __attribute__((ext_vector_type(8)))  __bf16 v8bf;
typedef __attribute__((ext_vector_type(8)))  float  v8f;
typedef unsigned int u32x4 __attribute__((ext_vector_type(4)));
typedef int          i32x4 __attribute__((ext_vector_type(4)));
typedef int          i32x8 __attribute__((ext_vector_type(8)));

#define N_AG      16384
#define OBS_LEN_  20
#define PRED_LEN_ 29
#define E_DIM     128
#define H_DIM     512
#define K_DIM     640          // E_DIM + H_DIM
#define N_GATE    2048         // 4*H
#define M_TILE    64           // agents per workgroup
#define KT_N      20           // K_DIM / 32
#define NT_N      128          // N_GATE / 16
#define THREADS   256          // 8 wave32 waves

// ---- LDS layout (dynamic shared; 291,328 B < 320 KB WGP LDS) ----
#define OFF_H0    0
#define OFF_H1    (OFF_H0 + M_TILE*H_DIM*2)           // h double buffer (bf16)
#define OFF_X     (OFF_H1 + M_TILE*H_DIM*2)           // x embed (bf16)
#define OFF_C     (OFF_X  + M_TILE*E_DIM*2)           // c state (f32)
#define OFF_REL   (OFF_C  + M_TILE*H_DIM*4)
#define OFF_POS   (OFF_REL + M_TILE*2*4)
#define OFF_OUT   (OFF_POS + M_TILE*2*4)
#define OFF_HIST  (OFF_OUT + M_TILE*5*4)
#define OFF_WPRED (OFF_HIST + M_TILE*4)               // W_pred staged via TDM (f32)
#define LDS_BYTES (OFF_WPRED + H_DIM*5*4)

#if defined(__has_builtin)
#if __has_builtin(__builtin_amdgcn_tensor_load_to_lds)
#define HAVE_TDM 1
#endif
#endif

__device__ __forceinline__ float sigf(float x) { return 1.0f / (1.0f + __expf(-x)); }
__device__ __forceinline__ float tanh_fast(float x) { return 2.0f * sigf(2.0f * x) - 1.0f; }

// -------------------------------------------------------------------------
// Pack W_ih (128x2048) and W_hh (512x2048) into bf16 B-fragment layout:
// wpack[((kt*128 + nt)*32 + lane)*16 + i] holds W[k][n] with
//   n = nt*16 + (lane&15),  k = kt*32 + (lane<16 ? 0 : 8) + (i<8 ? i : i+8)
// (ISA 7.12.2 dense 16-bit B layout: lanes 0-15 = K 0..7,16..23, lanes 16-31 = K 8..15,24..31)
// Also combines b_ih + b_hh into one f32 bias vector.
// -------------------------------------------------------------------------
__global__ void pack_weights(const float* __restrict__ W_ih,
                             const float* __restrict__ W_hh,
                             const float* __restrict__ b_ih,
                             const float* __restrict__ b_hh,
                             bf16* __restrict__ wpack,
                             float* __restrict__ bias) {
  int idx = blockIdx.x * blockDim.x + threadIdx.x;
  const int WELEMS = K_DIM * N_GATE;
  if (idx < WELEMS) {
    int i    = idx & 15;
    int lane = (idx >> 4) & 31;
    int blk  = idx >> 9;          // kt*NT_N + nt
    int nt   = blk % NT_N;
    int kt   = blk / NT_N;
    int n = nt * 16 + (lane & 15);
    int k = kt * 32 + ((lane < 16) ? 0 : 8) + (i < 8 ? i : i + 8);
    float v = (k < E_DIM) ? W_ih[k * N_GATE + n] : W_hh[(k - E_DIM) * N_GATE + n];
    wpack[idx] = (bf16)v;
  } else if (idx < WELEMS + N_GATE) {
    int n = idx - WELEMS;
    bias[n] = b_ih[n] + b_hh[n];
  }
}

// -------------------------------------------------------------------------
// One LSTM cell step for the workgroup's 64 agents.
// gates[64 x 2048] = [x(64x128) | h(64x512)] @ Wpack(640x2048) + bias
// Accumulators start at inline 0 (WMMA SRC2 literal-0 path); bias folded into
// the pointwise phase. All B loads are immediate offsets from one per-lane base.
// -------------------------------------------------------------------------
__device__ __forceinline__ void lstm_cell(char* smem,
                                          const bf16* __restrict__ hR,
                                          bf16* __restrict__ hW,
                                          const bf16* __restrict__ wpack,
                                          const float* __restrict__ bias,
                                          bool use_mask, int thresh) {
  const bf16* x_lds = (const bf16*)(smem + OFF_X);
  float*      c_lds = (float*)(smem + OFF_C);
  const int*  hist  = (const int*)(smem + OFF_HIST);

  const int lane = threadIdx.x & 31;
  const int wave = threadIdx.x >> 5;
  const int col  = lane & 15;
  const int ka   = (lane < 16) ? 0 : 8;     // A-fragment K half (ISA 7.12.2)

  for (int jj = 0; jj < 4; ++jj) {
    const int jt = wave + jj * 8;           // hidden tile 0..31
    // Single per-lane base; every load below is a compile-time immediate offset.
    //   elem(kt,g) = kt*65536 + g*16384 + jt*512 + lane*16
    const bf16* bbase = wpack + (size_t)jt * 512 + lane * 16;
    const bf16* abase_x = x_lds + (size_t)col * E_DIM + ka;
    const bf16* abase_h = hR + (size_t)col * H_DIM + ka;

    v8f acc[4][4] = {};                     // [row tile][gate i,f,g,o], start at 0

#pragma unroll
    for (int kt = 0; kt < KT_N; ++kt) {
      union { v16bf v; v8bf h[2]; } B[4];
#pragma unroll
      for (int g = 0; g < 4; ++g) {
        B[g].h[0] = *(const v8bf*)(bbase + kt * 65536 + g * 16384);
        B[g].h[1] = *(const v8bf*)(bbase + kt * 65536 + g * 16384 + 8);
      }
      if (kt + 1 < KT_N)   // L2->L0 prefetch of next K slab (global_prefetch_b8)
        __builtin_prefetch(bbase + (kt + 1) * 65536, 0, 3);

#pragma unroll
      for (int rt = 0; rt < 4; ++rt) {
        const bf16* ap = (kt < 4) ? (abase_x + rt * 16 * E_DIM + kt * 32)
                                  : (abase_h + rt * 16 * H_DIM + (kt - 4) * 32);
        union { v16bf v; v8bf h[2]; } A;
        A.h[0] = *(const v8bf*)(ap);        // K {base .. base+7}
        A.h[1] = *(const v8bf*)(ap + 16);   // K {base+16 .. base+23}
#pragma unroll
        for (int g = 0; g < 4; ++g)
          acc[rt][g] = __builtin_amdgcn_wmma_f32_16x16x32_bf16(
              false, A.v, false, B[g].v, (short)0, acc[rt][g], false, false);
      }
    }

    // ---- pointwise gate math + state update (C layout: VGPR v -> M=v/v+8) ----
    const float bvi = bias[0 * H_DIM + jt * 16 + col];
    const float bvf = bias[1 * H_DIM + jt * 16 + col];
    const float bvg = bias[2 * H_DIM + jt * 16 + col];
    const float bvo = bias[3 * H_DIM + jt * 16 + col];
#pragma unroll
    for (int rt = 0; rt < 4; ++rt) {
#pragma unroll
      for (int v = 0; v < 8; ++v) {
        const int m = rt * 16 + v + ((lane < 16) ? 0 : 8);
        const int j = jt * 16 + col;
        float gi = acc[rt][0][v] + bvi;
        float gf = acc[rt][1][v] + bvf;
        float gg = acc[rt][2][v] + bvg;
        float go = acc[rt][3][v] + bvo;
        float cold = c_lds[m * H_DIM + j];
        float c2 = sigf(gf) * cold + sigf(gi) * tanh_fast(gg);
        float h2 = sigf(go) * tanh_fast(c2);
        if (use_mask && !(hist[m] > thresh)) {
          c2 = cold;
          h2 = (float)hR[m * H_DIM + j];
        }
        c_lds[m * H_DIM + j] = c2;
        hW[m * H_DIM + j] = (bf16)h2;
      }
    }
  }
}

// -------------------------------------------------------------------------
// Persistent kernel: each block owns 64 agents for all 19 obs + 29 pred steps.
// -------------------------------------------------------------------------
__global__ void __launch_bounds__(THREADS)
lstm_persistent(const int* __restrict__ hist_g,
                const float* __restrict__ obs_pos,
                const float* __restrict__ obs_pos_rel,
                const float* __restrict__ W_embed,
                const float* __restrict__ b_embed,
                const float* __restrict__ W_pred,
                const float* __restrict__ b_pred,
                const float* __restrict__ h0,
                const bf16* __restrict__ wpack,
                const float* __restrict__ bias,
                float* __restrict__ out) {
  extern __shared__ char smem[];
  bf16*  hb0     = (bf16*)(smem + OFF_H0);
  bf16*  hb1     = (bf16*)(smem + OFF_H1);
  bf16*  x_lds   = (bf16*)(smem + OFF_X);
  float* c_lds   = (float*)(smem + OFF_C);
  float* rel_lds = (float*)(smem + OFF_REL);
  float* pos_lds = (float*)(smem + OFF_POS);
  float* out_lds = (float*)(smem + OFF_OUT);
  int*   hist_l  = (int*)(smem + OFF_HIST);
  float* wpred_l = (float*)(smem + OFF_WPRED);

  const int tid = threadIdx.x;
  const int agent0 = blockIdx.x * M_TILE;

#ifdef HAVE_TDM
  // ---- Stage W_pred (512x5 f32 = 10 KB) into LDS via the Tensor Data Mover.
  // 2D descriptor (D#): data_size=4B, tensor 2560x1, tile 2560x1. Issued once
  // by wave 0; completion tracked with TENSORcnt (ISA ch.8).
  if (tid < 32) {
    unsigned long long ga = (unsigned long long)(const void*)W_pred;
    unsigned lds_off = (unsigned)(unsigned long long)(void*)wpred_l; // LDS byte offset
    u32x4 g0;
    g0[0] = 1u;                                   // count=1, user mode
    g0[1] = lds_off;                              // lds_addr
    g0[2] = (unsigned)(ga & 0xFFFFFFFFull);       // global_addr[31:0]
    g0[3] = (unsigned)((ga >> 32) & 0x01FFFFFFull) | (2u << 30); // addr[56:32], type=2
    i32x8 g1;
    g1[0] = (int)(2u << 16);                      // data_size=4B, mask=0, no pad/iter
    g1[1] = (int)(2560u << 16);                   // tensor_dim0[15:0]=2560
    g1[2] = (int)(1u << 16);                      // tensor_dim0 hi=0, tensor_dim1=1
    g1[3] = (int)(2560u << 16);                   // tile_dim0=2560
    g1[4] = 1;                                    // tile_dim1=1, tile_dim2=0
    g1[5] = 2560;                                 // tensor_dim0_stride = 2560
    g1[6] = 0;
    g1[7] = 0;
    i32x4 z4 = {0, 0, 0, 0};
#if __clang_major__ >= 23
    i32x8 z8 = {0, 0, 0, 0, 0, 0, 0, 0};
    __builtin_amdgcn_tensor_load_to_lds(g0, g1, z4, z4, z8, 0);
#else
    __builtin_amdgcn_tensor_load_to_lds(g0, g1, z4, z4, 0);
#endif
  }
#else
  for (int idx = tid; idx < H_DIM * 5; idx += THREADS) wpred_l[idx] = W_pred[idx];
#endif

  // init h = c = tile(h0)
  for (int idx = tid; idx < M_TILE * H_DIM; idx += THREADS) {
    float v = h0[idx & (H_DIM - 1)];
    hb0[idx] = (bf16)v;
    c_lds[idx] = v;
  }
  if (tid < M_TILE) {
    int a = agent0 + tid;
    hist_l[tid] = hist_g[a];
    pos_lds[tid * 2 + 0] = obs_pos[(a * OBS_LEN_ + (OBS_LEN_ - 1)) * 2 + 0];
    pos_lds[tid * 2 + 1] = obs_pos[(a * OBS_LEN_ + (OBS_LEN_ - 1)) * 2 + 1];
  }
#ifdef HAVE_TDM
  if (tid < 32) {
#if __has_builtin(__builtin_amdgcn_s_wait_tensorcnt)
    __builtin_amdgcn_s_wait_tensorcnt(0);
#else
    asm volatile("s_wait_tensorcnt 0x0" ::: "memory");
#endif
  }
#endif
  __syncthreads();

  int cur = 0;

  // ---------------- observation phase: t = 1..19, masked updates ------------
  for (int t = 1; t < OBS_LEN_; ++t) {
    if (tid < M_TILE) {
      int a = agent0 + tid;
      rel_lds[tid * 2 + 0] = obs_pos_rel[(a * OBS_LEN_ + t) * 2 + 0];
      rel_lds[tid * 2 + 1] = obs_pos_rel[(a * OBS_LEN_ + t) * 2 + 1];
    }
    __syncthreads();
    for (int idx = tid; idx < M_TILE * E_DIM; idx += THREADS) {
      int m = idx >> 7, e = idx & (E_DIM - 1);
      float x = rel_lds[m * 2] * W_embed[e] + rel_lds[m * 2 + 1] * W_embed[E_DIM + e] + b_embed[e];
      x_lds[idx] = (bf16)fmaxf(x, 0.0f);
    }
    __syncthreads();
    lstm_cell(smem, cur ? hb1 : hb0, cur ? hb0 : hb1, wpack, bias, true, OBS_LEN_ - t);
    __syncthreads();
    cur ^= 1;
  }

  // ---------------- prediction phase: 29 steps ------------------------------
  for (int p = 0; p < PRED_LEN_; ++p) {
    const bf16* h = cur ? hb1 : hb0;
    // out = h @ W_pred + b_pred  (64x512 x 512x5), W_pred from LDS
    for (int task = tid; task < M_TILE * 5; task += THREADS) {
      int m = task / 5, q = task % 5;
      float s = b_pred[q];
      for (int k = 0; k < H_DIM; k += 8) {
        v8bf hv = *(const v8bf*)(h + m * H_DIM + k);
#pragma unroll
        for (int u = 0; u < 8; ++u) s += (float)hv[u] * wpred_l[(k + u) * 5 + q];
      }
      out_lds[task] = s;
      out[((size_t)(agent0 + m) * PRED_LEN_ + p) * 5 + q] = s;   // pred_out
    }
    __syncthreads();
    if (tid < M_TILE) {
      float r0 = out_lds[tid * 5 + 0], r1 = out_lds[tid * 5 + 1];
      float p0 = pos_lds[tid * 2 + 0] + r0, p1 = pos_lds[tid * 2 + 1] + r1;
      pos_lds[tid * 2 + 0] = p0; pos_lds[tid * 2 + 1] = p1;
      rel_lds[tid * 2 + 0] = r0; rel_lds[tid * 2 + 1] = r1;
      size_t toff = (size_t)N_AG * PRED_LEN_ * 5 +
                    ((size_t)(agent0 + tid) * PRED_LEN_ + p) * 2;
      out[toff + 0] = p0; out[toff + 1] = p1;                    // pred_traj
    }
    __syncthreads();
    if (p + 1 < PRED_LEN_) {   // last cell update's result is unused in reference
      for (int idx = tid; idx < M_TILE * E_DIM; idx += THREADS) {
        int m = idx >> 7, e = idx & (E_DIM - 1);
        float x = rel_lds[m * 2] * W_embed[e] + rel_lds[m * 2 + 1] * W_embed[E_DIM + e] + b_embed[e];
        x_lds[idx] = (bf16)fmaxf(x, 0.0f);
      }
      __syncthreads();
      lstm_cell(smem, cur ? hb1 : hb0, cur ? hb0 : hb1, wpack, bias, false, 0);
      __syncthreads();
      cur ^= 1;
    }
  }
}

extern "C" void kernel_launch(void* const* d_in, const int* in_sizes, int n_in,
                              void* d_out, int out_size, void* d_ws, size_t ws_size,
                              hipStream_t stream) {
  (void)in_sizes; (void)n_in; (void)out_size; (void)ws_size;
  const int*   hist    = (const int*)d_in[0];
  const float* obs_pos = (const float*)d_in[1];
  const float* obs_rel = (const float*)d_in[2];
  // d_in[3] lane_features, d_in[4] same_obs_mask: unused by reference outputs
  const float* W_embed = (const float*)d_in[5];
  const float* b_embed = (const float*)d_in[6];
  const float* W_ih    = (const float*)d_in[7];
  const float* W_hh    = (const float*)d_in[8];
  const float* b_ih    = (const float*)d_in[9];
  const float* b_hh    = (const float*)d_in[10];
  const float* W_pred  = (const float*)d_in[11];
  const float* b_pred  = (const float*)d_in[12];
  const float* h0      = (const float*)d_in[13];

  bf16*  wpack = (bf16*)d_ws;
  float* bias  = (float*)((char*)d_ws + (size_t)K_DIM * N_GATE * sizeof(bf16));

  const int packElems = K_DIM * N_GATE + N_GATE;
  pack_weights<<<(packElems + 255) / 256, 256, 0, stream>>>(W_ih, W_hh, b_ih, b_hh,
                                                            wpack, bias);

  lstm_persistent<<<N_AG / M_TILE, THREADS, LDS_BYTES, stream>>>(
      hist, obs_pos, obs_rel, W_embed, b_embed, W_pred, b_pred, h0,
      wpack, bias, (float*)d_out);
}
